// GCNNet_7421703488155
// MI455X (gfx1250) — matrix operand order repaired and checked by reading, not compiled, not run on hardware
//
#include <hip/hip_runtime.h>
#include <hip/hip_bf16.h>

typedef __attribute__((ext_vector_type(16))) _Float16 v16h;
typedef __attribute__((ext_vector_type(8)))  float    v8f;

// ---------------------------------------------------------------------------
// Degree / normalization kernels
// ---------------------------------------------------------------------------
__global__ void init_deg_kernel(float* __restrict__ deg, int N) {
    int i = blockIdx.x * blockDim.x + threadIdx.x;
    if (i < N) deg[i] = 1.0f;  // self loop
}

__global__ void deg_edges_kernel(const int* __restrict__ dst, float* __restrict__ deg, int E) {
    int e = blockIdx.x * blockDim.x + threadIdx.x;
    if (e < E) atomicAdd(&deg[dst[e]], 1.0f);
}

__global__ void dinv_kernel(float* __restrict__ dinv, int N) {
    int i = blockIdx.x * blockDim.x + threadIdx.x;
    if (i < N) dinv[i] = rsqrtf(dinv[i]);  // deg >= 1 always (self loop)
}

// ---------------------------------------------------------------------------
// GEMM1: xw1 = x @ W1   (N x 128) @ (128 x 32), f16 WMMA, f32 accum.
// Also initializes agg1 = dinv[row]^2 * xw1[row]   (self-loop contribution).
// One wave computes a 16-row stripe across both 16-col tiles (shared A frag).
// ---------------------------------------------------------------------------
__global__ void __launch_bounds__(256) gemm1_kernel(const float* __restrict__ x,
                                                    const float* __restrict__ W1,
                                                    const float* __restrict__ dinv,
                                                    float* __restrict__ xw1,
                                                    float* __restrict__ agg1,
                                                    int N) {
    // W1 is [128,32] row-major; store transposed f16 [n][k] for contiguous-K reads.
    __shared__ alignas(32) _Float16 Wt[32 * 128];
    for (int i = threadIdx.x; i < 32 * 128; i += 256) {
        int n = i >> 7, k = i & 127;
        Wt[i] = (_Float16)W1[k * 32 + n];
    }
    __syncthreads();

    const int wave = blockIdx.x * (blockDim.x >> 5) + (threadIdx.x >> 5);
    const int lane = threadIdx.x & 31;
    const int numM = (N + 15) >> 4;
    if (wave >= numM) return;  // wave-uniform: EXEC stays all-1s for WMMA

    const int row0 = wave << 4;
    const int rlo  = lane & 15;
    const int base = (lane < 16) ? 0 : 16;   // A frag K sub-base
    const int koff = (lane < 16) ? 0 : 16;   // B frag K sub-base
    int arow = row0 + rlo;
    if (arow >= N) arow = N - 1;             // clamp tail loads (stores are guarded)
    const float* __restrict__ xrow = x + (size_t)arow * 128;

    v8f c0 = {}; v8f c1 = {};
#pragma unroll
    for (int ks = 0; ks < 4; ++ks) {
        const int kb = ks * 32;
        // A fragment: K = kb + base + {0..15}, contiguous float4s
        const float4 a0 = *(const float4*)(xrow + kb + base);
        const float4 a1 = *(const float4*)(xrow + kb + base + 4);
        const float4 a2 = *(const float4*)(xrow + kb + base + 8);
        const float4 a3 = *(const float4*)(xrow + kb + base + 12);
        v16h A;
        A[0]=(_Float16)a0.x;  A[1]=(_Float16)a0.y;  A[2]=(_Float16)a0.z;  A[3]=(_Float16)a0.w;
        A[4]=(_Float16)a1.x;  A[5]=(_Float16)a1.y;  A[6]=(_Float16)a1.z;  A[7]=(_Float16)a1.w;
        A[8]=(_Float16)a2.x;  A[9]=(_Float16)a2.y;  A[10]=(_Float16)a2.z; A[11]=(_Float16)a2.w;
        A[12]=(_Float16)a3.x; A[13]=(_Float16)a3.y; A[14]=(_Float16)a3.z; A[15]=(_Float16)a3.w;
        // B fragments: col = rlo (+16), K = kb + koff + {0..15} contiguous in Wt
        const v16h B0 = *(const v16h*)&Wt[(size_t)rlo * 128 + kb + koff];
        const v16h B1 = *(const v16h*)&Wt[(size_t)(16 + rlo) * 128 + kb + koff];
        c0 = __builtin_amdgcn_wmma_f32_16x16x32_f16(false, A, false, B0, (short)0, c0, false, false);
        c1 = __builtin_amdgcn_wmma_f32_16x16x32_f16(false, A, false, B1, (short)0, c1, false, false);
    }

    const int rbase = (lane < 16) ? 0 : 8;  // D layout: VGPR r -> row r (+8 for hi lanes)
#pragma unroll
    for (int r = 0; r < 8; ++r) {
        const int row = row0 + rbase + r;
        if (row < N) {
            const float dv = dinv[row];
            const float sl = dv * dv;
            const size_t o = (size_t)row * 32 + rlo;
            xw1[o]       = c0[r];
            xw1[o + 16]  = c1[r];
            agg1[o]      = sl * c0[r];
            agg1[o + 16] = sl * c1[r];
        }
    }
}

// ---------------------------------------------------------------------------
// Edge gather/scale/scatter: 2^csh threads per edge, float4 chunk per thread.
// agg[dst] += dinv[src]*dinv[dst] * feat[src]
// ---------------------------------------------------------------------------
__global__ void scatter_edges_kernel(const int* __restrict__ src, const int* __restrict__ dst,
                                     const float* __restrict__ dinv,
                                     const float* __restrict__ feat,
                                     float* __restrict__ agg,
                                     int E, int F, int csh) {
    const long long t = (long long)blockIdx.x * blockDim.x + threadIdx.x;
    if (t >= ((long long)E << csh)) return;
    const int e = (int)(t >> csh);
    const int c = (int)(t & ((1 << csh) - 1)) << 2;
    const int s = src[e], d = dst[e];
    const float en = dinv[s] * dinv[d];
    const float4 v = *(const float4*)(feat + (size_t)s * F + c);
    float* p = agg + (size_t)d * F + c;
    atomicAdd(p + 0, en * v.x);
    atomicAdd(p + 1, en * v.y);
    atomicAdd(p + 2, en * v.z);
    atomicAdd(p + 3, en * v.w);
}

// ---------------------------------------------------------------------------
// Fused bias + ReLU + dropout(p=0.5) in place, deterministic stateless hash.
// ---------------------------------------------------------------------------
__global__ void relu_dropout_kernel(float* __restrict__ h, const float* __restrict__ b1,
                                    long long total) {
    const long long i = (long long)blockIdx.x * blockDim.x + threadIdx.x;
    if (i >= total) return;
    float v = h[i] + b1[(int)(i & 31)];
    v = fmaxf(v, 0.0f);
    unsigned hs = (unsigned)i ^ (42u * 0x9E3779B9u);
    hs *= 0x85EBCA6Bu; hs ^= hs >> 13; hs *= 0xC2B2AE35u; hs ^= hs >> 16;
    h[i] = (hs & 0x80000000u) ? v * 2.0f : 0.0f;
}

// ---------------------------------------------------------------------------
// GEMM2: hw2 = h @ W2  (N x 32) @ (32 x 16) — single K=32 WMMA per tile.
// Also initializes out = dinv^2 * hw2 + b2  (self-loop + bias fused).
// ---------------------------------------------------------------------------
__global__ void __launch_bounds__(256) gemm2_kernel(const float* __restrict__ h,
                                                    const float* __restrict__ W2,
                                                    const float* __restrict__ b2,
                                                    const float* __restrict__ dinv,
                                                    float* __restrict__ hw2,
                                                    float* __restrict__ out,
                                                    int N) {
    __shared__ alignas(32) _Float16 Wt[16 * 32];  // [n][k]
    for (int i = threadIdx.x; i < 16 * 32; i += 256) {
        int n = i >> 5, k = i & 31;
        Wt[i] = (_Float16)W2[k * 16 + n];
    }
    __syncthreads();

    const int wave = blockIdx.x * (blockDim.x >> 5) + (threadIdx.x >> 5);
    const int lane = threadIdx.x & 31;
    const int numM = (N + 15) >> 4;
    if (wave >= numM) return;

    const int row0 = wave << 4;
    const int rlo  = lane & 15;
    const int base = (lane < 16) ? 0 : 16;
    const int koff = (lane < 16) ? 0 : 16;
    int arow = row0 + rlo;
    if (arow >= N) arow = N - 1;
    const float* __restrict__ hrow = h + (size_t)arow * 32;

    const float4 a0 = *(const float4*)(hrow + base);
    const float4 a1 = *(const float4*)(hrow + base + 4);
    const float4 a2 = *(const float4*)(hrow + base + 8);
    const float4 a3 = *(const float4*)(hrow + base + 12);
    v16h A;
    A[0]=(_Float16)a0.x;  A[1]=(_Float16)a0.y;  A[2]=(_Float16)a0.z;  A[3]=(_Float16)a0.w;
    A[4]=(_Float16)a1.x;  A[5]=(_Float16)a1.y;  A[6]=(_Float16)a1.z;  A[7]=(_Float16)a1.w;
    A[8]=(_Float16)a2.x;  A[9]=(_Float16)a2.y;  A[10]=(_Float16)a2.z; A[11]=(_Float16)a2.w;
    A[12]=(_Float16)a3.x; A[13]=(_Float16)a3.y; A[14]=(_Float16)a3.z; A[15]=(_Float16)a3.w;

    const v16h B = *(const v16h*)&Wt[(size_t)rlo * 32 + koff];
    v8f c = {};
    c = __builtin_amdgcn_wmma_f32_16x16x32_f16(false, A, false, B, (short)0, c, false, false);

    const float bias = b2[rlo];
    const int rbase = (lane < 16) ? 0 : 8;
#pragma unroll
    for (int r = 0; r < 8; ++r) {
        const int row = row0 + rbase + r;
        if (row < N) {
            const float dv = dinv[row];
            const size_t o = (size_t)row * 16 + rlo;
            hw2[o] = c[r];
            out[o] = dv * dv * c[r] + bias;
        }
    }
}

// ---------------------------------------------------------------------------
extern "C" void kernel_launch(void* const* d_in, const int* in_sizes, int n_in,
                              void* d_out, int out_size, void* d_ws, size_t ws_size,
                              hipStream_t stream) {
    const float* x  = (const float*)d_in[0];
    const int*   ei = (const int*)d_in[1];
    const float* W1 = (const float*)d_in[2];
    const float* b1 = (const float*)d_in[3];
    const float* W2 = (const float*)d_in[4];
    const float* b2 = (const float*)d_in[5];
    float* out = (float*)d_out;

    const int N = in_sizes[0] / 128;
    const int E = in_sizes[1] / 2;
    const int* src = ei;
    const int* dst = ei + E;

    // Workspace layout (floats): dinv[N] | xw1[N*32] (reused as hw2[N*16]) | agg1[N*32]
    float* ws   = (float*)d_ws;
    float* dinv = ws;
    float* xw1  = ws + N;
    float* agg1 = ws + N + (size_t)N * 32;
    float* hw2  = xw1;  // xw1 dead after scatter1

    const int T = 256;
    // 1-3: symmetric normalization dinv = rsqrt(deg incl. self loop)
    init_deg_kernel<<<(N + T - 1) / T, T, 0, stream>>>(dinv, N);
    deg_edges_kernel<<<(E + T - 1) / T, T, 0, stream>>>(dst, dinv, E);
    dinv_kernel<<<(N + T - 1) / T, T, 0, stream>>>(dinv, N);

    // 4: xw1 = x@W1 (WMMA), agg1 = dinv^2 * xw1
    const int numM = (N + 15) / 16;
    gemm1_kernel<<<(numM + 7) / 8, T, 0, stream>>>(x, W1, dinv, xw1, agg1, N);

    // 5: layer-1 message scatter (8 threads/edge, 4 feats each)
    {
        long long work = (long long)E << 3;
        scatter_edges_kernel<<<(unsigned)((work + T - 1) / T), T, 0, stream>>>(
            src, dst, dinv, xw1, agg1, E, 32, 3);
    }

    // 6: h = dropout(relu(agg1 + b1)) in place
    {
        long long work = (long long)N * 32;
        relu_dropout_kernel<<<(unsigned)((work + T - 1) / T), T, 0, stream>>>(agg1, b1, work);
    }

    // 7: hw2 = h@W2 (WMMA), out = dinv^2 * hw2 + b2
    gemm2_kernel<<<(numM + 7) / 8, T, 0, stream>>>(agg1, W2, b2, dinv, hw2, out, N);

    // 8: layer-2 message scatter (4 threads/edge)
    {
        long long work = (long long)E << 2;
        scatter_edges_kernel<<<(unsigned)((work + T - 1) / T), T, 0, stream>>>(
            src, dst, dinv, hw2, out, E, 16, 2);
    }
}
